// GAT_74337293959507
// MI455X (gfx1250) — compile-verified, matrix operand order checked by reading, not run on hardware
//
#include <hip/hip_runtime.h>
#include <hip/hip_bf16.h>
#include <math.h>

#define N_NODES  50000
#define N_EDGES  500000
#define HIDDEN   64
#define HEADS    8
#define VOCABSZ  40000
#define N_GRAPHS 512

typedef __attribute__((ext_vector_type(16))) _Float16 v16h;
typedef __attribute__((ext_vector_type(4)))  _Float16 v4h;
typedef __attribute__((ext_vector_type(8)))  float    v8f;

// ---------------------------------------------------------------- utilities
__device__ __forceinline__ float atomic_max_f32(float* addr, float val) {
  if (val >= 0.0f)
    return __int_as_float(atomicMax((int*)addr, __float_as_int(val)));
  else
    return __uint_as_float(atomicMin((unsigned int*)addr, __float_as_uint(val)));
}

__global__ void fill_f32(float* __restrict__ p, float v, long long n) {
  long long i = (long long)blockIdx.x * blockDim.x + threadIdx.x;
  if (i < n) p[i] = v;
}

// ---------------------------------------------------------------- embedding
__global__ void embed_kernel(const int* __restrict__ x, const float* __restrict__ emb,
                             float* __restrict__ h, int n) {
  int i = blockIdx.x * blockDim.x + threadIdx.x;
  if (i >= n * HIDDEN) return;
  int node = i / HIDDEN, c = i - node * HIDDEN;
  h[i] = emb[(long long)x[node] * HIDDEN + c];
}

// ---------------------------------------------------------------- WMMA GEMM
// C[M,N] = A[M,K] @ B[K,N] (+ bias[N]); fp32 in/out, f16 WMMA 16x16x32 inside.
// Block = 256 threads = 8 waves (4 M x 2 N), block tile 128x64, K-step 32.
// LDS panels in WMMA fragment order (lane-major, 16 halfs / 32B per lane):
//   A 16x32 fragment: lane = r + ((k&15)>=8 ? 16:0), elem = (k>=16?8:0)+(k&7)
//   B 32x16 fragment: lane = c + ((k&15)>=8 ? 16:0), elem = (k>=16?8:0)+(k&7)
// Global side uses float4 (global_load_b128); A's 4 consecutive k-values map to
// one lane / 4 consecutive elems -> single ds_store_b64 of a packed v4h.
// Out-of-range M rows are CLAMPED (not branched): M%16==0 so partial tiles are
// tile-aligned; invalid tiles are simply never stored.
// Requires: N % 64 == 0, K % 32 == 0, M % 16 == 0 (true for all calls here).
#define GBM 128
#define GBN 64
#define GBK 32

__global__ __launch_bounds__(256) void gemm_f16_wmma(
    const float* __restrict__ A, const float* __restrict__ B,
    const float* __restrict__ bias, float* __restrict__ C,
    int M, int N, int K) {
  __shared__ _Float16 shA[8 * 32 * 16];   // 8 M-tiles x (32 lanes x 16 halfs)
  __shared__ _Float16 shB[4 * 32 * 16];   // 4 N-tiles x (32 lanes x 16 halfs)

  int tid    = threadIdx.x;
  int waveId = tid >> 5;
  int lane   = tid & 31;

  int nBlocksN = N / GBN;
  int bm = blockIdx.x / nBlocksN;
  int bn = blockIdx.x - bm * nBlocksN;
  int m0 = bm * GBM, n0 = bn * GBN;

  int wm = waveId >> 1;                    // 0..3  (M direction, 2 tiles each)
  int wn = waveId & 1;                     // 0..1  (N direction, 2 tiles each)

  v8f acc00 = {}, acc01 = {}, acc10 = {}, acc11 = {};

  for (int kk = 0; kk < K; kk += GBK) {
    // ---- A panel: 128x32 f32 = 1024 float4; 4 vectors / thread, clamped rows
    float4 av[4];
#pragma unroll
    for (int it = 0; it < 4; ++it) {
      int j   = it * 256 + tid;            // vector index 0..1023
      int row = j >> 3;                    // 8 float4 per row
      int k4  = (j & 7) << 2;
      int gr  = m0 + row;
      if (gr >= M) gr = M - 1;             // clamp: no branchy loads
      av[it] = *(const float4*)&A[(long long)gr * K + kk + k4];
    }
    // ---- B panel: 32x64 f32 = 512 float4; 2 vectors / thread
    float4 bv[2];
#pragma unroll
    for (int it = 0; it < 2; ++it) {
      int j  = it * 256 + tid;             // vector index 0..511
      int k  = j >> 4;                     // 16 float4 per k-row
      int n4 = (j & 15) << 2;
      bv[it] = *(const float4*)&B[(long long)(kk + k) * N + n0 + n4];
    }
    // ---- convert + scatter to LDS fragment order
#pragma unroll
    for (int it = 0; it < 4; ++it) {
      int j   = it * 256 + tid;
      int row = j >> 3;
      int k4  = (j & 7) << 2;
      int t   = row >> 4, r = row & 15;
      int ln  = r + (((k4 & 15) >= 8) ? 16 : 0);
      int e0  = ((k4 >= 16) ? 8 : 0) + (k4 & 7);
      v4h p;
      p[0] = (_Float16)av[it].x; p[1] = (_Float16)av[it].y;
      p[2] = (_Float16)av[it].z; p[3] = (_Float16)av[it].w;
      *(v4h*)&shA[(t * 32 + ln) * 16 + e0] = p;        // one ds_store_b64
    }
#pragma unroll
    for (int it = 0; it < 2; ++it) {
      int j  = it * 256 + tid;
      int k  = j >> 4;
      int n4 = (j & 15) << 2;
      int t  = n4 >> 4;
      int e  = ((k >= 16) ? 8 : 0) + (k & 7);
      int lb = (n4 & 15) + (((k & 15) >= 8) ? 16 : 0); // 4 consecutive lanes
      shB[(t * 32 + lb + 0) * 16 + e] = (_Float16)bv[it].x;
      shB[(t * 32 + lb + 1) * 16 + e] = (_Float16)bv[it].y;
      shB[(t * 32 + lb + 2) * 16 + e] = (_Float16)bv[it].z;
      shB[(t * 32 + lb + 3) * 16 + e] = (_Float16)bv[it].w;
    }
    __syncthreads();

    // ---- fragment loads (contiguous 32B per lane) + 2x2 WMMA
    v16h a0 = *(const v16h*)&shA[((wm * 2 + 0) * 32 + lane) * 16];
    v16h a1 = *(const v16h*)&shA[((wm * 2 + 1) * 32 + lane) * 16];
    v16h b0 = *(const v16h*)&shB[((wn * 2 + 0) * 32 + lane) * 16];
    v16h b1 = *(const v16h*)&shB[((wn * 2 + 1) * 32 + lane) * 16];

    acc00 = __builtin_amdgcn_wmma_f32_16x16x32_f16(false, a0, false, b0, (short)0, acc00, false, false);
    acc01 = __builtin_amdgcn_wmma_f32_16x16x32_f16(false, a0, false, b1, (short)0, acc01, false, false);
    acc10 = __builtin_amdgcn_wmma_f32_16x16x32_f16(false, a1, false, b0, (short)0, acc10, false, false);
    acc11 = __builtin_amdgcn_wmma_f32_16x16x32_f16(false, a1, false, b1, (short)0, acc11, false, false);

    __syncthreads();
  }

  // ---- store: C element (i) of acc maps to M = i + (lane<16?0:8), N = lane&15
  int col = lane & 15;
  int mb  = (lane < 16) ? 0 : 8;
#pragma unroll
  for (int i = 0; i < 2; ++i) {
    int tileM = m0 + (wm * 2 + i) * 16;
    if (tileM >= M) continue;              // wave-uniform guard (M % 16 == 0)
#pragma unroll
    for (int j = 0; j < 2; ++j) {
      int n = n0 + (wn * 2 + j) * 16 + col;
      float bv = bias ? bias[n] : 0.0f;
      v8f acc = (i == 0) ? ((j == 0) ? acc00 : acc01)
                         : ((j == 0) ? acc10 : acc11);
#pragma unroll
      for (int r = 0; r < 8; ++r)
        C[(long long)(tileM + mb + r) * N + n] = acc[r] + bv;
    }
  }
}

// ---------------------------------------------------------------- attention
__global__ void node_scores(const float* __restrict__ hp, const float* __restrict__ a_src,
                            const float* __restrict__ a_dst, float* __restrict__ s_src,
                            float* __restrict__ s_dst, int n, int H) {
  int idx = blockIdx.x * blockDim.x + threadIdx.x;
  if (idx >= n * H) return;
  int node = idx / H, h = idx - node * H;
  const float* v = hp + (long long)node * H * HIDDEN + h * HIDDEN;
  float ss = 0.f, sd = 0.f;
#pragma unroll 8
  for (int c = 0; c < HIDDEN; ++c) {
    float xv = v[c];
    ss += xv * a_src[h * HIDDEN + c];
    sd += xv * a_dst[h * HIDDEN + c];
  }
  s_src[idx] = ss;
  s_dst[idx] = sd;
}

__device__ __forceinline__ void edge_nodes(const int* __restrict__ ei, int e,
                                           int* s, int* d) {
  if (e < N_EDGES) { *s = ei[e]; *d = ei[N_EDGES + e]; }
  else             { *s = e - N_EDGES; *d = e - N_EDGES; }   // self-loops
}

__device__ __forceinline__ float leaky02(float l) { return (l > 0.f) ? l : 0.2f * l; }

__global__ void edge_max(const int* __restrict__ ei, const float* __restrict__ s_src,
                         const float* __restrict__ s_dst, float* __restrict__ mbuf, int H) {
  int idx = blockIdx.x * blockDim.x + threadIdx.x;
  int EE = N_EDGES + N_NODES;
  if (idx >= EE * H) return;
  int e = idx / H, h = idx - e * H;
  int s, d; edge_nodes(ei, e, &s, &d);
  float l = leaky02(s_src[s * H + h] + s_dst[d * H + h]);
  atomic_max_f32(&mbuf[d * H + h], l);
}

__global__ void edge_expsum(const int* __restrict__ ei, const float* __restrict__ s_src,
                            const float* __restrict__ s_dst, const float* __restrict__ mbuf,
                            float* __restrict__ dbuf, int H) {
  int idx = blockIdx.x * blockDim.x + threadIdx.x;
  int EE = N_EDGES + N_NODES;
  if (idx >= EE * H) return;
  int e = idx / H, h = idx - e * H;
  int s, d; edge_nodes(ei, e, &s, &d);
  float l = leaky02(s_src[s * H + h] + s_dst[d * H + h]);
  atomicAdd(&dbuf[d * H + h], __expf(l - mbuf[d * H + h]));
}

// one wave per (edge, head); lanes stream the 64 channels (2 per lane)
__global__ void edge_msg(const int* __restrict__ ei, const float* __restrict__ s_src,
                         const float* __restrict__ s_dst, const float* __restrict__ mbuf,
                         const float* __restrict__ dbuf, const float* __restrict__ hp,
                         float* __restrict__ agg, int H) {
  int gid  = blockIdx.x * blockDim.x + threadIdx.x;
  int wave = gid >> 5;
  int lane = gid & 31;
  int EE = N_EDGES + N_NODES;
  if (wave >= EE * H) return;
  int e = wave / H, h = wave - e * H;
  int s, d; edge_nodes(ei, e, &s, &d);
  float l = leaky02(s_src[s * H + h] + s_dst[d * H + h]);
  float alpha = __expf(l - mbuf[d * H + h]) / (dbuf[d * H + h] + 1e-16f);
  const float* sv = hp  + (long long)s * H * HIDDEN + h * HIDDEN;
  float*       dv = agg + (long long)d * H * HIDDEN + h * HIDDEN;
#pragma unroll
  for (int c = lane; c < HIDDEN; c += 32)
    atomicAdd(&dv[c], alpha * sv[c]);
}

// ---------------------------------------------------------------- pointwise
__global__ void bias_act(float* __restrict__ x, const float* __restrict__ b,
                         long long total, int F, int elu) {
  long long i = (long long)blockIdx.x * blockDim.x + threadIdx.x;
  if (i >= total) return;
  float v = x[i] + b[i % F];
  if (elu) v = (v > 0.f) ? v : (__expf(v) - 1.f);
  x[i] = v;
}

__global__ void pool_kernel(const float* __restrict__ h2, const int* __restrict__ batch,
                            float* __restrict__ pooled, int n) {
  int i = blockIdx.x * blockDim.x + threadIdx.x;
  if (i >= n * HIDDEN) return;
  int node = i / HIDDEN, c = i - node * HIDDEN;
  atomicAdd(&pooled[batch[node] * HIDDEN + c], h2[i]);
}

// ---------------------------------------------------------------- launcher
static inline int blocks_for(long long threads, int bs) {
  return (int)((threads + bs - 1) / bs);
}

static inline void launch_gemm(const float* A, const float* B, const float* bias,
                               float* C, int M, int N, int K, hipStream_t stream) {
  int gm = (M + GBM - 1) / GBM;
  int gn = N / GBN;
  gemm_f16_wmma<<<gm * gn, 256, 0, stream>>>(A, B, bias, C, M, N, K);
}

extern "C" void kernel_launch(void* const* d_in, const int* in_sizes, int n_in,
                              void* d_out, int out_size, void* d_ws, size_t ws_size,
                              hipStream_t stream) {
  (void)in_sizes; (void)n_in; (void)out_size; (void)ws_size;
  const int*   x     = (const int*)d_in[0];
  const int*   ei    = (const int*)d_in[1];
  const int*   batch = (const int*)d_in[2];
  const float* emb   = (const float*)d_in[3];
  const float* W1    = (const float*)d_in[4];
  const float* a_s1  = (const float*)d_in[5];
  const float* a_d1  = (const float*)d_in[6];
  const float* b1    = (const float*)d_in[7];
  const float* W2    = (const float*)d_in[8];
  const float* a_s2  = (const float*)d_in[9];
  const float* a_d2  = (const float*)d_in[10];
  const float* b2    = (const float*)d_in[11];
  const float* Wp    = (const float*)d_in[12];
  const float* bp    = (const float*)d_in[13];
  float* out = (float*)d_out;

  // workspace layout (bytes), 256B aligned, with reuse
  char* ws = (char*)d_ws;
  size_t off = 0;
  auto alloc = [&](size_t bytes) {
    char* p = ws + off; off = (off + bytes + 255) & ~(size_t)255; return p;
  };
  float* h0     = (float*)alloc((size_t)N_NODES * HIDDEN * 4);           // emb out; later hp2
  float* hp     = (float*)alloc((size_t)N_NODES * HEADS * HIDDEN * 4);   // layer1 features; later agg2
  float* agg1   = (float*)alloc((size_t)N_NODES * HEADS * HIDDEN * 4);   // layer1 aggregation -> h1
  float* ssrc   = (float*)alloc((size_t)N_NODES * HEADS * 4);
  float* sdst   = (float*)alloc((size_t)N_NODES * HEADS * 4);
  float* mbuf   = (float*)alloc((size_t)N_NODES * HEADS * 4);
  float* dbuf   = (float*)alloc((size_t)N_NODES * HEADS * 4);
  float* pooled = (float*)alloc((size_t)N_GRAPHS * HIDDEN * 4);

  const int BS = 256;
  const int EE = N_EDGES + N_NODES;
  const float NEG_HUGE = -3.0e38f;

  // 1) embedding gather
  embed_kernel<<<blocks_for((long long)N_NODES * HIDDEN, BS), BS, 0, stream>>>(x, emb, h0, N_NODES);

  // 2) GEMM1: hp = h0 @ W1   [50000,64] @ [64,512]
  launch_gemm(h0, W1, nullptr, hp, N_NODES, HEADS * HIDDEN, HIDDEN, stream);

  // 3) layer-1 attention scores + segment softmax + aggregation
  node_scores<<<blocks_for((long long)N_NODES * HEADS, BS), BS, 0, stream>>>(
      hp, a_s1, a_d1, ssrc, sdst, N_NODES, HEADS);
  fill_f32<<<blocks_for((long long)N_NODES * HEADS, BS), BS, 0, stream>>>(
      mbuf, NEG_HUGE, (long long)N_NODES * HEADS);
  fill_f32<<<blocks_for((long long)N_NODES * HEADS, BS), BS, 0, stream>>>(
      dbuf, 0.f, (long long)N_NODES * HEADS);
  fill_f32<<<blocks_for((long long)N_NODES * HEADS * HIDDEN, BS), BS, 0, stream>>>(
      agg1, 0.f, (long long)N_NODES * HEADS * HIDDEN);
  edge_max<<<blocks_for((long long)EE * HEADS, BS), BS, 0, stream>>>(ei, ssrc, sdst, mbuf, HEADS);
  edge_expsum<<<blocks_for((long long)EE * HEADS, BS), BS, 0, stream>>>(ei, ssrc, sdst, mbuf, dbuf, HEADS);
  edge_msg<<<blocks_for((long long)EE * HEADS * 32, BS), BS, 0, stream>>>(
      ei, ssrc, sdst, mbuf, dbuf, hp, agg1, HEADS);

  // 4) h1 = elu(agg1 + b1)
  bias_act<<<blocks_for((long long)N_NODES * HEADS * HIDDEN, BS), BS, 0, stream>>>(
      agg1, b1, (long long)N_NODES * HEADS * HIDDEN, HEADS * HIDDEN, 1);

  // 5) GEMM2: hp2 = h1 @ W2   [50000,512] @ [512,64]  (hp2 reuses h0 region)
  float* hp2 = h0;
  launch_gemm(agg1, W2, nullptr, hp2, N_NODES, HIDDEN, HEADS * HIDDEN, stream);

  // 6) layer-2 attention (1 head); agg2 reuses hp region
  float* agg2 = hp;
  node_scores<<<blocks_for((long long)N_NODES, BS), BS, 0, stream>>>(
      hp2, a_s2, a_d2, ssrc, sdst, N_NODES, 1);
  fill_f32<<<blocks_for((long long)N_NODES, BS), BS, 0, stream>>>(mbuf, NEG_HUGE, (long long)N_NODES);
  fill_f32<<<blocks_for((long long)N_NODES, BS), BS, 0, stream>>>(dbuf, 0.f, (long long)N_NODES);
  fill_f32<<<blocks_for((long long)N_NODES * HIDDEN, BS), BS, 0, stream>>>(
      agg2, 0.f, (long long)N_NODES * HIDDEN);
  edge_max<<<blocks_for((long long)EE, BS), BS, 0, stream>>>(ei, ssrc, sdst, mbuf, 1);
  edge_expsum<<<blocks_for((long long)EE, BS), BS, 0, stream>>>(ei, ssrc, sdst, mbuf, dbuf, 1);
  edge_msg<<<blocks_for((long long)EE * 32, BS), BS, 0, stream>>>(
      ei, ssrc, sdst, mbuf, dbuf, hp2, agg2, 1);

  // 7) h2 = agg2 + b2 (no activation)
  bias_act<<<blocks_for((long long)N_NODES * HIDDEN, BS), BS, 0, stream>>>(
      agg2, b2, (long long)N_NODES * HIDDEN, HIDDEN, 0);

  // 8) global add pool
  fill_f32<<<blocks_for((long long)N_GRAPHS * HIDDEN, BS), BS, 0, stream>>>(
      pooled, 0.f, (long long)N_GRAPHS * HIDDEN);
  pool_kernel<<<blocks_for((long long)N_NODES * HIDDEN, BS), BS, 0, stream>>>(
      agg2, batch, pooled, N_NODES);

  // 9) GEMM3: out = pooled @ Wp + bp   [512,64] @ [64,40000]
  launch_gemm(pooled, Wp, bp, out, N_GRAPHS, VOCABSZ, HIDDEN, stream);
}
